// SparseMoE_77163382440675
// MI455X (gfx1250) — matrix operand order repaired and checked by reading, not compiled, not run on hardware
//
#include <hip/hip_runtime.h>
#include <hip/hip_bf16.h>

// ---------------- problem constants ----------------
#define Bm   4
#define Sm   2048
#define BS   (Bm * Sm)     // 8192 tokens
#define Dm   512
#define Hm   1024
#define Em   8
#define TOPK 2
#define MT   32            // tokens per MoE block (two 16-row WMMA M-tiles)

typedef __attribute__((ext_vector_type(16))) __bf16 v16bf;
typedef __attribute__((ext_vector_type(8)))  float  v8f;
typedef __attribute__((ext_vector_type(4)))  unsigned int v4u;

union Frag32B { v16bf bf; v4u u[2]; };

// float -> bf16 (round-to-nearest-even)
__device__ __forceinline__ unsigned short f2bf(float f) {
    union { float f; unsigned int u; } v; v.f = f;
    unsigned int u = v.u;
    unsigned int r = u + 0x7FFFu + ((u >> 16) & 1u);
    return (unsigned short)(r >> 16);
}

// A-fragment: 16x32 bf16, row-major [M][K] source with row stride ld (ushorts).
__device__ __forceinline__ v16bf load_a_frag(const unsigned short* base, int m,
                                             int sel, int kbase, int ld) {
    const unsigned short* p0 = base + m * ld + kbase + sel * 8;
    Frag32B f;
    f.u[0] = *(const v4u*)(p0);
    f.u[1] = *(const v4u*)(p0 + 16);
    return f.bf;
}

// B-fragment: 32x16 bf16; source row-major [N][K] (pre-transposed weight).
__device__ __forceinline__ v16bf load_b_frag(const unsigned short* brow,
                                             int sel, int kbase) {
    const unsigned short* p0 = brow + kbase + sel * 8;
    Frag32B f;
    f.u[0] = *(const v4u*)(p0);
    f.u[1] = *(const v4u*)(p0 + 16);
    return f.bf;
}

__device__ __forceinline__ v8f wmma_bf16(v16bf a, v16bf b, v8f c) {
    return __builtin_amdgcn_wmma_f32_16x16x32_bf16(false, a, false, b,
                                                   (short)0, c, false, false);
}

// ---------------- kernel 1: zero outputs + counters ----------------
__global__ void zero_kernel(float* outF, int n, float* usage, int* cursors) {
    int i = blockIdx.x * blockDim.x + threadIdx.x;
    for (int j = i; j < n; j += gridDim.x * blockDim.x) outF[j] = 0.0f;
    if (i < Em) { usage[i] = 0.0f; cursors[i] = 0; }
}

// ---------------- kernel 2: tiled f32 [E][R][C] -> bf16 [E][C][R] ----------------
__global__ void __launch_bounds__(256)
cvt_transpose_kernel(const float* __restrict__ in,
                     unsigned short* __restrict__ out, int R, int C) {
    __shared__ unsigned short tile[32][33];
    int e  = blockIdx.z;
    int c0 = blockIdx.x * 32, r0 = blockIdx.y * 32;
    int tx = threadIdx.x & 31, ty = threadIdx.x >> 5;   // 32 x 8
#pragma unroll
    for (int i = 0; i < 32; i += 8) {
        int r = r0 + ty + i, c = c0 + tx;
        tile[ty + i][tx] = f2bf(in[((long)e * R + r) * C + c]);
    }
    __syncthreads();
#pragma unroll
    for (int i = 0; i < 32; i += 8) {
        int c = c0 + ty + i, r = r0 + tx;
        out[((long)e * C + c) * R + r] = tile[tx][ty + i];
    }
}

// ---------------- kernel 3: routing ----------------
__global__ void __launch_bounds__(256)
routing_kernel(const float* __restrict__ x, const float* __restrict__ noise,
               const float* __restrict__ gw, const float* __restrict__ gb,
               unsigned short* __restrict__ xn, float* usage, int* cursors,
               int* tokList, float* probList, int* outIdx) {
    int t = blockIdx.x;
    const float* xt = x + (long)t * Dm;
    __shared__ float red[256];
    __shared__ float lg[Em];
    int tid = threadIdx.x;

    red[tid] = xt[tid] + xt[tid + 256];
    __syncthreads();
    for (int off = 128; off; off >>= 1) {
        if (tid < off) red[tid] += red[tid + off];
        __syncthreads();
    }
    float total = red[0];
    float mean  = total * (1.0f / (float)Dm);
    bool  keep  = mean > 0.1f;                       // SPIKE_THRESHOLD
    float inv   = keep ? 1.0f / (total + 1e-8f) : 0.0f;

    xn[(long)t * Dm + tid]       = f2bf(xt[tid] * inv);
    xn[(long)t * Dm + tid + 256] = f2bf(xt[tid + 256] * inv);

    int wave = tid >> 5, lane = tid & 31;
    float acc = 0.0f;
    for (int d = lane; d < Dm; d += 32)
        acc += (xt[d] * inv) * gw[d * Em + wave];
    for (int off = 16; off; off >>= 1) acc += __shfl_down(acc, off, 32);
    if (lane == 0) lg[wave] = acc;
    __syncthreads();

    if (tid == 0) {
        float l[Em];
        for (int e = 0; e < Em; ++e)
            l[e] = lg[e] + gb[e] + noise[(long)t * Em + e] * 0.01f;
        int e0 = 0;
        for (int e = 1; e < Em; ++e) if (l[e] > l[e0]) e0 = e;
        int e1 = (e0 == 0) ? 1 : 0;
        for (int e = 0; e < Em; ++e) if (e != e0 && l[e] > l[e1]) e1 = e;
        float thr = l[e1], mx = l[e0];
        float p[Em], psum = 0.0f;
        for (int e = 0; e < Em; ++e) {
            float me = (l[e] >= thr) ? l[e] : -1e9f;
            p[e] = expf(me - mx);
            psum += p[e];
        }
        float ip = 1.0f / psum;
        for (int e = 0; e < Em; ++e) { p[e] *= ip; atomicAdd(usage + e, p[e]); }
        outIdx[t * 2]     = e0;
        outIdx[t * 2 + 1] = e1;
        int pos0 = atomicAdd(cursors + e0, 1);
        tokList[e0 * BS + pos0]  = t;
        probList[e0 * BS + pos0] = p[e0];
        int pos1 = atomicAdd(cursors + e1, 1);
        tokList[e1 * BS + pos1]  = t;
        probList[e1 * BS + pos1] = p[e1];
    }
}

// ---------------- kernel 4: aux loss ----------------
__global__ void aux_kernel(const float* usage, float* outAux) {
    if (threadIdx.x == 0 && blockIdx.x == 0) {
        float s = 0.0f;
        for (int e = 0; e < Em; ++e) s += usage[e];
        float inv = 1.0f / s, m = 1.0f / (float)Em, var = 0.0f;
        for (int e = 0; e < Em; ++e) {
            float d = usage[e] * inv - m;
            var += d * d;
        }
        var *= (1.0f / (float)Em);
        outAux[0] = sqrtf(var) / (m + 1e-10f);
    }
}

// ---------------- kernel 5: WMMA expert MLP over gathered tokens ----------------
// Block (tile, e): MT=32 tokens of expert e. 512 threads = 16 waves.
// Loop-structure rationale (from rounds 1-5 codegen):
//  - K loops ROLLED with a x4 partial unroll and NO manual rotation: this was
//    the only variant with zero scratch spills AND zero register-rotation
//    moves (manual double-buffering lowered to ~16 v_mov + hazard nops/iter).
//  - nt loops pinned to unroll 1 so the compiler cannot replicate the K loop
//    and LICM-promote all nt-invariant A fragments (256 VGPRs -> spills).
//  - Latency hiding comes from occupancy: ~100 VGPRs/wave leaves 8+ resident
//    waves per SIMD to cover the short LDS/L2 waits.
__global__ void __launch_bounds__(512)
moe_kernel(const unsigned short* __restrict__ xn,
           const unsigned short* __restrict__ w1T,   // bf16 [E][H][D]
           const unsigned short* __restrict__ w2T,   // bf16 [E][D][H]
           const float* __restrict__ b1, const float* __restrict__ b2,
           const int* __restrict__ cursors, const int* __restrict__ tokList,
           const float* __restrict__ probList, float* outF) {
    int e    = blockIdx.y;
    int tile = blockIdx.x;
    int count = cursors[e];
    if (tile * MT >= count) return;

    __shared__ __attribute__((aligned(16))) unsigned short sX[MT * Dm]; // 32 KB
    __shared__ __attribute__((aligned(16))) unsigned short sH[MT * Hm]; // 64 KB
    __shared__ int   sTok[MT];
    __shared__ float sP[MT];

    int tid = threadIdx.x;
    if (tid < MT) {
        int idx  = tile * MT + tid;
        int safe = (idx < count) ? idx : (count - 1);
        sTok[tid] = tokList[e * BS + safe];
        sP[tid]   = (idx < count) ? probList[e * BS + idx] : 0.0f;
    }
    __syncthreads();

    // Gathered token rows -> LDS via CDNA5 async global->LDS copies.
    for (int c = tid; c < MT * (Dm / 8); c += 512) {
        int row = c >> 6;
        int off = (c & 63) << 3;
        unsigned int ldsa =
            (unsigned int)(size_t)(sX + row * Dm + off);          // LDS byte addr
        unsigned long long ga =
            (unsigned long long)(size_t)(xn + (long)sTok[row] * Dm + off);
        asm volatile("global_load_async_to_lds_b128 %0, %1, off"
                     :: "v"(ldsa), "v"(ga) : "memory");
    }
    asm volatile("s_wait_asynccnt 0x0" ::: "memory");
    __syncthreads();

    int wave = tid >> 5, lane = tid & 31;
    int nl = lane & 15, sel = lane >> 4;

    // ---- GEMM1: x @ w1 + b1, silu -> sH ----
    const unsigned short* w1e = w1T + (long)e * Hm * Dm;
#pragma unroll 1
    for (int nt = wave * 4; nt < wave * 4 + 4; ++nt) {
        int nbase = nt * 16;
        v8f acc0 = {}, acc1 = {};
        const unsigned short* brow = w1e + (long)(nbase + nl) * Dm;
#pragma unroll 4
        for (int ks = 0; ks < Dm / 32; ++ks) {
            v16bf b  = load_b_frag(brow, sel, ks * 32);
            v16bf a0 = load_a_frag(sX,           nl, sel, ks * 32, Dm);
            v16bf a1 = load_a_frag(sX + 16 * Dm, nl, sel, ks * 32, Dm);
            acc0 = wmma_bf16(a0, b, acc0);
            acc1 = wmma_bf16(a1, b, acc1);
        }

        float b1v = b1[e * Hm + nbase + nl];
#pragma unroll
        for (int r = 0; r < 8; ++r) {
            float h0 = acc0[r] + b1v;
            float h1 = acc1[r] + b1v;
            h0 = h0 / (1.0f + expf(-h0));          // silu
            h1 = h1 / (1.0f + expf(-h1));
            sH[(r + 8 * sel) * Hm + nbase + nl]      = f2bf(h0);
            sH[(16 + r + 8 * sel) * Hm + nbase + nl] = f2bf(h1);
        }
    }
    __syncthreads();

    // ---- GEMM2: h @ w2 + b2, weighted scatter-add ----
    const unsigned short* w2e = w2T + (long)e * Dm * Hm;
#pragma unroll 1
    for (int nt = wave * 2; nt < wave * 2 + 2; ++nt) {
        int nbase = nt * 16;
        v8f acc0 = {}, acc1 = {};
        const unsigned short* brow = w2e + (long)(nbase + nl) * Hm;
#pragma unroll 4
        for (int ks = 0; ks < Hm / 32; ++ks) {
            v16bf b  = load_b_frag(brow, sel, ks * 32);
            v16bf a0 = load_a_frag(sH,           nl, sel, ks * 32, Hm);
            v16bf a1 = load_a_frag(sH + 16 * Hm, nl, sel, ks * 32, Hm);
            acc0 = wmma_bf16(a0, b, acc0);
            acc1 = wmma_bf16(a1, b, acc1);
        }

        float b2v = b2[e * Dm + nbase + nl];
#pragma unroll
        for (int r = 0; r < 8; ++r) {
            int m0 = r + 8 * sel;
            int m1 = 16 + r + 8 * sel;
            atomicAdd(outF + (long)sTok[m0] * Dm + nbase + nl,
                      sP[m0] * (acc0[r] + b2v));
            atomicAdd(outF + (long)sTok[m1] * Dm + nbase + nl,
                      sP[m1] * (acc1[r] + b2v));
        }
    }
}

// ---------------- host launcher ----------------
extern "C" void kernel_launch(void* const* d_in, const int* in_sizes, int n_in,
                              void* d_out, int out_size, void* d_ws, size_t ws_size,
                              hipStream_t stream) {
    (void)in_sizes; (void)n_in; (void)out_size;
    const float* x     = (const float*)d_in[0];
    const float* noise = (const float*)d_in[1];
    const float* gw    = (const float*)d_in[2];
    const float* gb    = (const float*)d_in[3];
    const float* w1    = (const float*)d_in[4];
    const float* b1    = (const float*)d_in[5];
    const float* w2    = (const float*)d_in[6];
    const float* b2    = (const float*)d_in[7];

    float* outF   = (float*)d_out;                       // [B,S,D] f32
    int*   outIdx = (int*)(outF + (size_t)BS * Dm);      // [B,S,2] i32
    float* outAux = outF + (size_t)BS * Dm + (size_t)BS * TOPK; // scalar

    char* ws = (char*)d_ws;
    size_t off = 0;
    unsigned short* xn  = (unsigned short*)(ws + off); off += (size_t)BS * Dm * 2;      // 8 MB
    unsigned short* w1T = (unsigned short*)(ws + off); off += (size_t)Em * Hm * Dm * 2; // 8 MB
    unsigned short* w2T = (unsigned short*)(ws + off); off += (size_t)Em * Dm * Hm * 2; // 8 MB
    float* usage   = (float*)(ws + off); off += Em * sizeof(float);
    int*   cursors = (int*)(ws + off);   off += Em * sizeof(int);
    int*   tokList = (int*)(ws + off);   off += (size_t)Em * BS * sizeof(int);          // 256 KB
    float* probList= (float*)(ws + off); off += (size_t)Em * BS * sizeof(float);        // 256 KB
    (void)ws_size;

    zero_kernel<<<4096, 256, 0, stream>>>(outF, BS * Dm, usage, cursors);

    // w1 [E][D][H] -> w1T [E][H][D];  w2 [E][H][D] -> w2T [E][D][H]
    cvt_transpose_kernel<<<dim3(Hm / 32, Dm / 32, Em), 256, 0, stream>>>(w1, w1T, Dm, Hm);
    cvt_transpose_kernel<<<dim3(Dm / 32, Hm / 32, Em), 256, 0, stream>>>(w2, w2T, Hm, Dm);

    routing_kernel<<<BS, 256, 0, stream>>>(x, noise, gw, gb, xn, usage, cursors,
                                           tokList, probList, outIdx);
    aux_kernel<<<1, 32, 0, stream>>>(usage, outAux);

    dim3 grid(BS / MT, Em);
    moe_kernel<<<grid, 512, 0, stream>>>(xn, w1T, w2T, b1, b2,
                                         cursors, tokList, probList, outF);
}
